// ScaleAttentionMechanism_8521215115892
// MI455X (gfx1250) — matrix operand order repaired and checked by reading, not compiled, not run on hardware
//
#include <hip/hip_runtime.h>
#include <hip/hip_bf16.h>
#include <math.h>

typedef __attribute__((ext_vector_type(16))) _Float16 v16h;
typedef __attribute__((ext_vector_type(8)))  float    v8f;

#define BATCH   4
#define CH      128
#define KOFF    27
#define PPART   64          // partial blocks per batch for stats
#define BN_EPS  1e-5f

// ---------------- Stage 1a: per-batch partial sum/max over feats ----------------
__global__ void stats_partial_kernel(const float* __restrict__ feats,
                                     float* __restrict__ psum,
                                     float* __restrict__ pmax, int nper) {
    int b = blockIdx.x / PPART;
    int part = blockIdx.x % PPART;
    int c = threadIdx.x;                       // 128 threads: one per channel
    int chunk = (nper + PPART - 1) / PPART;
    int p0 = part * chunk;
    int p1 = p0 + chunk; if (p1 > nper) p1 = nper;
    size_t base = ((size_t)b * nper) * CH;
    float s = 0.f, m = -3.402823e38f;
    for (int p = p0; p < p1; ++p) {
        float v = feats[base + (size_t)p * CH + c];
        s += v; m = fmaxf(m, v);
    }
    psum[(b * PPART + part) * CH + c] = s;
    pmax[(b * PPART + part) * CH + c] = m;
}

// ---------------- Stage 1b: finish reduction + channel-attention MLP ----------------
__global__ void weights_kernel(const float* __restrict__ psum,
                               const float* __restrict__ pmax,
                               const float* __restrict__ w1, const float* __restrict__ b1,
                               const float* __restrict__ w2, const float* __restrict__ b2,
                               float* __restrict__ weights, int nper) {
    int b = blockIdx.x;
    int c = threadIdx.x;                       // 128 threads
    __shared__ float avg[CH], mxv[CH], ha[CH / 2], hm[CH / 2];
    float s = 0.f, m = -3.402823e38f;
    for (int p = 0; p < PPART; ++p) {
        s += psum[(b * PPART + p) * CH + c];
        m = fmaxf(m, pmax[(b * PPART + p) * CH + c]);
    }
    avg[c] = s / (float)nper;
    mxv[c] = m;
    __syncthreads();
    if (c < CH / 2) {                          // hidden layer (64)
        float sa = b1[c], sm = b1[c];
        for (int cc = 0; cc < CH; ++cc) {
            float w = w1[cc * (CH / 2) + c];
            sa += avg[cc] * w; sm += mxv[cc] * w;
        }
        ha[c] = fmaxf(sa, 0.f); hm[c] = fmaxf(sm, 0.f);
    }
    __syncthreads();
    float oa = b2[c], om = b2[c];
    for (int j = 0; j < CH / 2; ++j) {
        float w = w2[j * CH + c];
        oa += ha[j] * w; om += hm[j] * w;
    }
    float z = oa + om;
    weights[b * CH + c] = 1.f / (1.f + expf(-z));
}

// ---------------- Stage 2a: fuse mask into gather index (masked -> zero row N) ----------------
__global__ void fuse_idx_kernel(const int* __restrict__ nbr_idx,
                                const unsigned char* __restrict__ nbr_mask,
                                int* __restrict__ idxm, int total, int N) {
    int f = blockIdx.x * blockDim.x + threadIdx.x;
    if (f >= total) return;
    idxm[f] = nbr_mask[f] ? nbr_idx[f] : N;    // row N of mid_h is all zeros
}

// ---------------- Stage 2b: mid = feats * weights[batch], stored as f16 ----------------
__global__ void mid_kernel(const float* __restrict__ feats,
                           const float* __restrict__ weights,
                           const int* __restrict__ batch_idx,
                           _Float16* __restrict__ mid_h, int N) {
    int t = blockIdx.x * blockDim.x + threadIdx.x;   // one thread per 8 channels
    int i = t >> 4;
    if (i >= N) return;
    int c8 = (t & 15) * 8;
    int b = batch_idx[i];
    const float4* f4 = (const float4*)(feats + (size_t)i * CH + c8);
    const float4* w4 = (const float4*)(weights + b * CH + c8);
    float4 f0 = f4[0], f1 = f4[1], w0 = w4[0], w1v = w4[1];
    union { uint4 u; _Float16 h[8]; } o;
    o.h[0] = (_Float16)(f0.x * w0.x);  o.h[1] = (_Float16)(f0.y * w0.y);
    o.h[2] = (_Float16)(f0.z * w0.z);  o.h[3] = (_Float16)(f0.w * w0.w);
    o.h[4] = (_Float16)(f1.x * w1v.x); o.h[5] = (_Float16)(f1.y * w1v.y);
    o.h[6] = (_Float16)(f1.z * w1v.z); o.h[7] = (_Float16)(f1.w * w1v.w);
    *((uint4*)(mid_h + (size_t)i * CH + c8)) = o.u;
}

// ---------------- Stage 3a: pack conv1_w into WMMA B-fragment layout ----------------
// Bws[((k*4+q)*32 + lane)*16 + e]: lane<16 -> col=lane, K rows {0..7,16..23} of chunk q;
// lane>=16 -> col=lane-16, K rows {8..15,24..31}. Columns 4..15 are zero.
__global__ void prep_b_kernel(const float* __restrict__ conv1_w, _Float16* __restrict__ Bws) {
    int f = blockIdx.x * blockDim.x + threadIdx.x;
    if (f >= KOFF * 4 * 32 * 16) return;
    int e = f & 15;
    int lane = (f >> 4) & 31;
    int q = (f >> 9) & 3;
    int k = f >> 11;
    int col = lane & 15;
    int row = q * 32 + ((lane >> 4) << 3) + (e < 8 ? e : e + 8);       // input channel 0..127
    float v = (col < 4) ? conv1_w[(k * CH + row) * 4 + col] : 0.f;
    Bws[f] = (_Float16)v;
}

// ---------------- Stage 3b: sparse conv via WMMA. One wave = 4 tiles (64 points) ----------------
// B fragment loads amortized over 4 accumulators; masked gathers hit the zero row.
__global__ __launch_bounds__(256)
void conv_wmma_kernel(const _Float16* __restrict__ mid_h,
                      const _Float16* __restrict__ Bws,
                      const int* __restrict__ idxm,
                      float* __restrict__ y_ws, int ntiles, int N) {
    int grp = blockIdx.x * 8 + (threadIdx.x >> 5);   // group of 4 consecutive tiles
    int ngroups = (ntiles + 3) >> 2;
    if (grp >= ngroups) return;                      // wave-uniform
    const int lane = threadIdx.x & 31;
    const int row = lane & 15;                       // matrix row within a tile
    const int off = (lane < 16) ? 0 : 8;             // K sub-offset per A layout
    const int i0 = grp * 64;

    int p[4];
#pragma unroll
    for (int t = 0; t < 4; ++t) {
        int pp = i0 + t * 16 + row;
        p[t] = (pp < N) ? pp : (N - 1);              // clamp (tail safety)
    }

    v8f acc[4] = {{}, {}, {}, {}};
    union AF { v16h h; uint4 u[2]; };

    for (int k = 0; k < KOFF; ++k) {
        const _Float16* rp[4];
#pragma unroll
        for (int t = 0; t < 4; ++t) {
            int idx = idxm[(size_t)p[t] * KOFF + k];
            rp[t] = mid_h + (size_t)idx * CH + off;
        }
        const _Float16* bp = Bws + (size_t)k * 2048 + lane * 16;
#pragma unroll
        for (int q = 0; q < 4; ++q) {
            v16h b = *(const v16h*)(bp + q * 512);
#pragma unroll
            for (int t = 0; t < 4; ++t) {
                // A fragment: lane<16 holds K {0..7,16..23} of its row; lane>=16 holds {8..15,24..31}
                AF a;
                a.u[0] = *(const uint4*)(rp[t] + q * 32);
                a.u[1] = *(const uint4*)(rp[t] + q * 32 + 16);
                acc[t] = __builtin_amdgcn_wmma_f32_16x16x32_f16(
                    false, a.h, false, b, (short)0, acc[t], false, false);
            }
        }
    }
    // C layout: VGPR r -> M=r (lanes 0..15, N=lane) / M=r+8 (lanes 16..31, N=lane-16)
    if ((lane & 15) < 4) {
        int m0 = (lane < 16) ? 0 : 8;
        int col = lane & 15;
#pragma unroll
        for (int t = 0; t < 4; ++t) {
            int base = i0 + t * 16;
            if (base < N) {
#pragma unroll
                for (int r = 0; r < 8; ++r)
                    y_ws[(size_t)(base + m0 + r) * 4 + col] = acc[t][r];
            }
        }
    }
}

// ---------------- Stage 4: BN statistics (sum, sumsq per channel) ----------------
__global__ void bn_stats_kernel(const float* __restrict__ y_ws, float* __restrict__ stats, int N) {
    float s[4] = {0, 0, 0, 0}, ss[4] = {0, 0, 0, 0};
    for (int i = blockIdx.x * blockDim.x + threadIdx.x; i < N; i += gridDim.x * blockDim.x) {
        float4 v = ((const float4*)y_ws)[i];
        s[0] += v.x; ss[0] += v.x * v.x;
        s[1] += v.y; ss[1] += v.y * v.y;
        s[2] += v.z; ss[2] += v.z * v.z;
        s[3] += v.w; ss[3] += v.w * v.w;
    }
    __shared__ float sm[8 * 256];
    int tid = threadIdx.x;
#pragma unroll
    for (int c = 0; c < 4; ++c) { sm[c * 256 + tid] = s[c]; sm[(c + 4) * 256 + tid] = ss[c]; }
    __syncthreads();
    for (int o = 128; o > 0; o >>= 1) {
        if (tid < o)
#pragma unroll
            for (int j = 0; j < 8; ++j) sm[j * 256 + tid] += sm[j * 256 + tid + o];
        __syncthreads();
    }
    if (tid < 8) atomicAdd(&stats[tid], sm[tid * 256]);
}

// ---------------- Stage 5: BN + ReLU + 4x4 linear + sigmoid + max -> gate ----------------
__global__ void gate_kernel(const float* __restrict__ y_ws, const float* __restrict__ stats,
                            const float* __restrict__ gamma, const float* __restrict__ beta,
                            const float* __restrict__ conv2_w, float* __restrict__ gate, int N) {
    int i = blockIdx.x * blockDim.x + threadIdx.x;
    if (i >= N) return;
    float invN = 1.f / (float)N;
    float4 yv = ((const float4*)y_ws)[i];
    float ya[4] = {yv.x, yv.y, yv.z, yv.w};
    float yb[4];
#pragma unroll
    for (int n = 0; n < 4; ++n) {
        float mu  = stats[n] * invN;
        float var = stats[4 + n] * invN - mu * mu;
        yb[n] = fmaxf((ya[n] - mu) * rsqrtf(var + BN_EPS) * gamma[n] + beta[n], 0.f);
    }
    float g = -3.402823e38f;
#pragma unroll
    for (int n = 0; n < 4; ++n) {
        float z = 0.f;
#pragma unroll
        for (int m = 0; m < 4; ++m) z += yb[m] * conv2_w[m * 4 + n];
        g = fmaxf(g, 1.f / (1.f + expf(-z)));
    }
    gate[i] = g;
}

// ---------------- Stage 6: out = mid*(gate+1) + feats ----------------
__global__ void out_kernel(const float* __restrict__ feats, const float* __restrict__ weights,
                           const float* __restrict__ gate, const int* __restrict__ batch_idx,
                           float* __restrict__ out, int N) {
    int t = blockIdx.x * blockDim.x + threadIdx.x;   // one thread per 4 channels
    int i = t >> 5;
    if (i >= N) return;
    int c4 = t & 31;
    float g = gate[i] + 1.f;
    int b = batch_idx[i];
    float4 f = ((const float4*)feats)[(size_t)i * 32 + c4];
    float4 w = ((const float4*)(weights + b * CH))[c4];
    float4 o;
    o.x = f.x * w.x * g + f.x;
    o.y = f.y * w.y * g + f.y;
    o.z = f.z * w.z * g + f.z;
    o.w = f.w * w.w * g + f.w;
    ((float4*)out)[(size_t)i * 32 + c4] = o;
}

extern "C" void kernel_launch(void* const* d_in, const int* in_sizes, int n_in,
                              void* d_out, int out_size, void* d_ws, size_t ws_size,
                              hipStream_t stream) {
    const float* feats    = (const float*)d_in[0];
    const float* w1       = (const float*)d_in[1];
    const float* b1       = (const float*)d_in[2];
    const float* w2       = (const float*)d_in[3];
    const float* b2       = (const float*)d_in[4];
    const float* conv1_w  = (const float*)d_in[5];
    const float* bn_gamma = (const float*)d_in[6];
    const float* bn_beta  = (const float*)d_in[7];
    const float* conv2_w  = (const float*)d_in[8];
    const int*   nbr_idx  = (const int*)d_in[9];
    const unsigned char* nbr_mask = (const unsigned char*)d_in[10];
    const int*   batch_idx = (const int*)d_in[11];
    float* out = (float*)d_out;

    const int N    = in_sizes[0] / CH;     // 200000
    const int nper = N / BATCH;            // 50000
    const int ntiles = (N + 15) / 16;
    const int nmap = N * KOFF;

    // ---- workspace carve-up (all offsets 256B aligned) ----
    char* w = (char*)d_ws;
    float*    stats   = (float*)(w + 0);                        // 8 f32
    float*    weights = (float*)(w + 256);                      // 512 f32
    float*    psum    = (float*)(w + 2304);                     // B*PPART*CH
    float*    pmax    = (float*)(w + 2304 + 131072);
    _Float16* Bws     = (_Float16*)(w + 264448);                // 27*4*512 f16
    size_t o = 264448 + (size_t)KOFF * 4 * 512 * 2;             // 375040
    float*    gate    = (float*)(w + o);  o += ((size_t)N * 4 + 255) & ~(size_t)255;
    float*    y_ws    = (float*)(w + o);  o += ((size_t)N * 16 + 255) & ~(size_t)255;
    _Float16* mid_h   = (_Float16*)(w + o); o += ((size_t)(N + 1) * CH * 2 + 255) & ~(size_t)255;
    int*      idxm    = (int*)(w + o);

    hipMemsetAsync(stats, 0, 8 * sizeof(float), stream);
    hipMemsetAsync(mid_h + (size_t)N * CH, 0, CH * sizeof(_Float16), stream); // zero row N

    stats_partial_kernel<<<BATCH * PPART, CH, 0, stream>>>(feats, psum, pmax, nper);
    weights_kernel<<<BATCH, CH, 0, stream>>>(psum, pmax, w1, b1, w2, b2, weights, nper);
    fuse_idx_kernel<<<(nmap + 255) / 256, 256, 0, stream>>>(nbr_idx, nbr_mask, idxm, nmap, N);
    prep_b_kernel<<<(KOFF * 4 * 512 + 255) / 256, 256, 0, stream>>>(conv1_w, Bws);
    mid_kernel<<<(N * 16 + 255) / 256, 256, 0, stream>>>(feats, weights, batch_idx, mid_h, N);
    int ngroups = (ntiles + 3) / 4;
    conv_wmma_kernel<<<(ngroups + 7) / 8, 256, 0, stream>>>(mid_h, Bws, idxm, y_ws, ntiles, N);
    bn_stats_kernel<<<256, 256, 0, stream>>>(y_ws, stats, N);
    gate_kernel<<<(N + 255) / 256, 256, 0, stream>>>(y_ws, stats, bn_gamma, bn_beta, conv2_w, gate, N);
    out_kernel<<<(N * 32 + 255) / 256, 256, 0, stream>>>(feats, weights, gate, batch_idx, out, N);
}
